// GPTModel_53360673685781
// MI455X (gfx1250) — matrix operand order repaired and checked by reading, not compiled, not run on hardware
//
#include <hip/hip_runtime.h>
#include <hip/hip_bf16.h>
#include <stdint.h>

// ---------------------------------------------------------------------------
// Mini-GPT forward for MI455X (gfx1250): bf16 WMMA GEMMs + WMMA flash attention
// ---------------------------------------------------------------------------

typedef __bf16 bf16;
typedef __attribute__((ext_vector_type(16))) __bf16 v16bf;
typedef __attribute__((ext_vector_type(8)))  float  v8f;
typedef __attribute__((ext_vector_type(4)))  int    v4i;

union Frag16 { v16bf v; unsigned int u[8]; bf16 h[16]; };
union FragAcc { v8f v; float f[8]; };

__device__ __forceinline__ v8f wmma_bf16f32(v16bf a, v16bf b, v8f c) {
  // D = A(16x32 bf16) * B(32x16 bf16) + C(16x16 f32)
  return __builtin_amdgcn_wmma_f32_16x16x32_bf16(false, a, false, b, (short)0, c,
                                                 false, false);
}

// CDNA5 async global->LDS copy path (ASYNCcnt), with sync fallback.
#if defined(__gfx1250__) && __has_builtin(__builtin_amdgcn_global_load_async_to_lds_b128)
#define ASYNC_LDS 1
__device__ __forceinline__ void async_copy16(const void* g, void* l) {
  __builtin_amdgcn_global_load_async_to_lds_b128(
      (__attribute__((address_space(1))) v4i*)(g),
      (__attribute__((address_space(3))) v4i*)(l), 0, 0);
}
__device__ __forceinline__ void wait_async() {
#if __has_builtin(__builtin_amdgcn_s_wait_asynccnt)
  __builtin_amdgcn_s_wait_asynccnt(0);
#else
  asm volatile("s_wait_asynccnt 0" ::: "memory");
#endif
}
#else
__device__ __forceinline__ void wait_async() {}
#endif

#define MTOK 32768     // B*T
#define CDIM 384
#define TSEQ 256
#define NH   6
#define DHD  64
#define VOC  65
#define NL   6

// ---------------------------------------------------------------------------
// Weight packing / conversion
// ---------------------------------------------------------------------------
__global__ void k_convert_bf16(const float* __restrict__ s, bf16* __restrict__ d, int n) {
  int i = blockIdx.x * blockDim.x + threadIdx.x;
  if (i < n) d[i] = (bf16)s[i];
}

// Wq/Wk/Wv [L,H,C,DH] -> fused bf16 [L][C][3C], columns ordered q|k|v head-major
__global__ void k_pack_qkv(const float* __restrict__ Wq, const float* __restrict__ Wk,
                           const float* __restrict__ Wv, bf16* __restrict__ out, int n) {
  int i = blockIdx.x * blockDim.x + threadIdx.x;
  if (i >= n) return;
  int l   = i / (CDIM * 3 * CDIM);
  int rem = i - l * (CDIM * 3 * CDIM);
  int c   = rem / (3 * CDIM);
  int col = rem - c * (3 * CDIM);
  int which = col / CDIM;
  int hc    = col - which * CDIM;
  int h = hc / DHD, d = hc - h * DHD;
  const float* W = (which == 0) ? Wq : ((which == 1) ? Wk : Wv);
  out[i] = (bf16)W[(((size_t)l * NH + h) * CDIM + c) * DHD + d];
}

// ---------------------------------------------------------------------------
// Embedding: x[b,t,c] = tok_emb[idx[b,t],c] + pos_emb[t,c]
// ---------------------------------------------------------------------------
__global__ void k_embed(const int* __restrict__ idx, const float* __restrict__ tok,
                        const float* __restrict__ pos, float* __restrict__ x) {
  int i = blockIdx.x * blockDim.x + threadIdx.x;
  if (i >= MTOK * CDIM) return;
  int m = i / CDIM, c = i - m * CDIM;
  int t = m & (TSEQ - 1);
  x[i] = tok[idx[m] * CDIM + c] + pos[t * CDIM + c];
}

// ---------------------------------------------------------------------------
// LayerNorm: one row per block of 128 threads (C=384 -> 3 elems/thread)
// ---------------------------------------------------------------------------
__global__ __launch_bounds__(128)
void k_layernorm(const float* __restrict__ x, const float* __restrict__ g,
                 const float* __restrict__ bta, float* __restrict__ outf,
                 bf16* __restrict__ outb) {
  __shared__ float r1[128], r2[128];
  const int m = blockIdx.x, tid = threadIdx.x;
  const float* xr = x + (size_t)m * CDIM;
  float v[3], s = 0.f, q = 0.f;
#pragma unroll
  for (int i = 0; i < 3; ++i) { v[i] = xr[tid + i * 128]; s += v[i]; q += v[i] * v[i]; }
  r1[tid] = s; r2[tid] = q;
  __syncthreads();
  for (int st = 64; st > 0; st >>= 1) {
    if (tid < st) { r1[tid] += r1[tid + st]; r2[tid] += r2[tid + st]; }
    __syncthreads();
  }
  const float mean = r1[0] * (1.0f / CDIM);
  const float var  = r2[0] * (1.0f / CDIM) - mean * mean;
  const float rstd = rsqrtf(var + 1e-5f);
#pragma unroll
  for (int i = 0; i < 3; ++i) {
    int c = tid + i * 128;
    float y = (v[i] - mean) * rstd * g[c] + bta[c];
    outb[(size_t)m * CDIM + c] = (bf16)y;
    if (outf) outf[(size_t)m * CDIM + c] = y;
  }
}

// ---------------------------------------------------------------------------
// Generic bf16 WMMA GEMM:  out[M,N] = A[M,K] * Bw[K,N] (+bias)(+resid)(relu)
// Block tile 256x64, 8 waves; each wave owns 32 rows x 64 cols:
//   2 A-fragments, 4 shared B-fragments, 8 WMMAs per 32-wide K chunk.
// Double-buffered LDS; A tile staged with async global->LDS when available.
// ---------------------------------------------------------------------------
#define BM 256
#define BN 64
#define BK 32

__global__ __launch_bounds__(256)
void k_gemm_bf16(const bf16* __restrict__ A, const bf16* __restrict__ Bw,
                 const float* __restrict__ bias, const float* __restrict__ resid,
                 float* __restrict__ outf, bf16* __restrict__ outb,
                 int M, int N, int K, int relu) {
  __shared__ bf16 As[2][BM][BK];    // 2 x 16 KB
  __shared__ bf16 Bst[2][BN][BK];   // 2 x 4 KB, transposed: [n][k]
  const int tid = threadIdx.x;
  const int wave = tid >> 5, lane = tid & 31;
  const int ln = lane & 15, hi = lane >> 4;
  const int m0 = blockIdx.x * BM;
  const int n0 = blockIdx.y * BN;

  FragAcc acc[2][4];
#pragma unroll
  for (int ms = 0; ms < 2; ++ms)
#pragma unroll
    for (int i = 0; i < 4; ++i)
#pragma unroll
      for (int j = 0; j < 8; ++j) acc[ms][i].f[j] = 0.0f;

  const int nk = K >> 5;

  auto stageA = [&](int kc, int buf) {
    // one 32-elem (64B) row per thread, 4x 16B segments
    const bf16* gsrc = A + (size_t)(m0 + tid) * K + (kc << 5);
#if defined(ASYNC_LDS)
#pragma unroll
    for (int i = 0; i < 4; ++i)
      async_copy16(gsrc + i * 8, &As[buf][tid][i * 8]);
#else
#pragma unroll
    for (int i = 0; i < 4; ++i)
      *reinterpret_cast<uint4*>(&As[buf][tid][i * 8]) =
          *reinterpret_cast<const uint4*>(gsrc + i * 8);
#endif
  };
  auto stageB = [&](int kc, int buf) {
    int row = tid >> 3, seg = tid & 7;   // k-row, 8-col segment
    uint4 t = *reinterpret_cast<const uint4*>(
        Bw + (size_t)((kc << 5) + row) * N + n0 + seg * 8);
    const bf16* hv = reinterpret_cast<const bf16*>(&t);
#pragma unroll
    for (int j = 0; j < 8; ++j) Bst[buf][seg * 8 + j][row] = hv[j];
  };

  stageA(0, 0);
  stageB(0, 0);
  for (int kc = 0; kc < nk; ++kc) {
    const int buf = kc & 1;
    wait_async();
    __syncthreads();              // tile[buf] ready; tile[buf^1] fully consumed
    if (kc + 1 < nk) {            // prefetch next chunk into the other buffer
      stageA(kc + 1, buf ^ 1);
      stageB(kc + 1, buf ^ 1);
    }
    if (kc + 2 < nk)
      __builtin_prefetch(A + (size_t)(m0 + tid) * K + ((kc + 2) << 5), 0, 1);

    // load ALL fragments, then burst 8 WMMAs
    Frag16 af[2], bfr[4];
#pragma unroll
    for (int ms = 0; ms < 2; ++ms)
#pragma unroll
      for (int p = 0; p < 8; ++p) {
        // A layout: element e -> K = e + 16*hi
        int k2 = 2 * p + hi * 16;
        af[ms].u[p] = *reinterpret_cast<const unsigned int*>(
            &As[buf][wave * 32 + ms * 16 + ln][k2]);
      }
#pragma unroll
    for (int nt = 0; nt < 4; ++nt)
#pragma unroll
      for (int p = 0; p < 8; ++p)   // B layout: element e -> K = e + 16*hi, N = ln
        bfr[nt].u[p] = *reinterpret_cast<const unsigned int*>(
            &Bst[buf][nt * 16 + ln][2 * p + hi * 16]);
#pragma unroll
    for (int ms = 0; ms < 2; ++ms)
#pragma unroll
      for (int nt = 0; nt < 4; ++nt)
        acc[ms][nt].v = wmma_bf16f32(af[ms].v, bfr[nt].v, acc[ms][nt].v);
  }

  // epilogue (C/D layout: VGPR r -> row r + 8*hi, col = ln)
#pragma unroll
  for (int ms = 0; ms < 2; ++ms)
#pragma unroll
    for (int nt = 0; nt < 4; ++nt) {
      int col = n0 + nt * 16 + ln;
      float bv = bias ? bias[col] : 0.0f;
#pragma unroll
      for (int r = 0; r < 8; ++r) {
        int row = m0 + wave * 32 + ms * 16 + r + hi * 8;
        float v = acc[ms][nt].f[r] + bv;
        if (resid) v += resid[(size_t)row * N + col];
        if (relu)  v = fmaxf(v, 0.0f);
        if (outf)  outf[(size_t)row * N + col] = v;
        if (outb)  outb[(size_t)row * N + col] = (bf16)v;
      }
    }
}

// ---------------------------------------------------------------------------
// Flash attention (causal) with WMMA. One block = one (b,h).
// qkv bf16 [B,T,1152] (q|k|v head-concat). 8 waves x 32 query rows.
// ---------------------------------------------------------------------------
__global__ __launch_bounds__(256)
void k_attention(const bf16* __restrict__ qkv, bf16* __restrict__ att) {
  __shared__ bf16 Vt[DHD][TSEQ];      // V transposed, 32 KB
  __shared__ bf16 Plds[8][16][32];    // per-wave P tile staging, 8 KB
  const int bh = blockIdx.x;
  const int b = bh / NH, h = bh % NH;
  const int tid = threadIdx.x, wave = tid >> 5, lane = tid & 31;
  const int ln = lane & 15, hi = lane >> 4;
  const size_t base = (size_t)b * TSEQ * (3 * CDIM);
  const bf16* Qp = qkv + base + h * DHD;
  const bf16* Kp = qkv + base + CDIM + h * DHD;
  const bf16* Vp = qkv + base + 2 * CDIM + h * DHD;

  // stage V transposed: Vt[d][t]
#pragma unroll
  for (int i = 0; i < 8; ++i) {
    int id2 = tid * 8 + i;
    int t = id2 >> 3, seg = id2 & 7;
    uint4 v = *reinterpret_cast<const uint4*>(Vp + (size_t)t * (3 * CDIM) + seg * 8);
    const bf16* hv = reinterpret_cast<const bf16*>(&v);
#pragma unroll
    for (int j = 0; j < 8; ++j) Vt[seg * 8 + j][t] = hv[j];
  }
  __syncthreads();

  const float scale = 0.125f;   // DH^-0.5
  const float NEG = -1.0e30f;

  for (int mt = 0; mt < 2; ++mt) {
    const int rowbase = wave * 32 + mt * 16;
    // Q A-fragments, 2 chunks of K(d)=32, loaded straight from global
    Frag16 qa[2];
    {
      const bf16* qr = Qp + (size_t)(rowbase + ln) * (3 * CDIM);
#pragma unroll
      for (int c = 0; c < 2; ++c)
#pragma unroll
        for (int p = 0; p < 8; ++p) {
          int d = c * 32 + 2 * p + hi * 16;
          qa[c].u[p] = *reinterpret_cast<const unsigned int*>(qr + d);
        }
    }
    FragAcc o[4];
#pragma unroll
    for (int i = 0; i < 4; ++i)
#pragma unroll
      for (int j = 0; j < 8; ++j) o[i].f[j] = 0.0f;
    float mrun[8], lsum[8];
#pragma unroll
    for (int r = 0; r < 8; ++r) { mrun[r] = NEG; lsum[r] = 0.0f; }

    const int nchunks = (rowbase + 16 + 31) >> 5;   // 32 keys per chunk
    for (int jc = 0; jc < nchunks; ++jc) {
      const int j0 = jc << 5;
      // preload all 4 K B-fragments, then burst the S WMMAs
      Frag16 kb[2][2];
#pragma unroll
      for (int sub = 0; sub < 2; ++sub) {
        const bf16* kr = Kp + (size_t)(j0 + sub * 16 + ln) * (3 * CDIM);
#pragma unroll
        for (int c = 0; c < 2; ++c) {  // element e -> d = c*32 + e + 16*hi
          const bf16* kp = kr + c * 32 + hi * 16;
          *reinterpret_cast<uint4*>(&kb[sub][c].u[0]) =
              *reinterpret_cast<const uint4*>(kp);
          *reinterpret_cast<uint4*>(&kb[sub][c].u[4]) =
              *reinterpret_cast<const uint4*>(kp + 8);
        }
      }
      FragAcc s[2];
#pragma unroll
      for (int sub = 0; sub < 2; ++sub) {
#pragma unroll
        for (int j = 0; j < 8; ++j) s[sub].f[j] = 0.0f;
#pragma unroll
        for (int c = 0; c < 2; ++c)
          s[sub].v = wmma_bf16f32(qa[c].v, kb[sub][c].v, s[sub].v);
      }
      // online softmax (rows live in VGPR index r + 8*hi; cols across 16 lanes)
#pragma unroll
      for (int r = 0; r < 8; ++r) {
        const int trow = rowbase + r + hi * 8;
        float a0 = (j0 + ln      <= trow) ? s[0].f[r] * scale : NEG;
        float a1 = (j0 + 16 + ln <= trow) ? s[1].f[r] * scale : NEG;
        float mx = fmaxf(a0, a1);
#pragma unroll
        for (int off = 1; off < 16; off <<= 1)
          mx = fmaxf(mx, __shfl_xor(mx, off, 32));
        const float newm = fmaxf(mrun[r], mx);
        const float corr = __expf(mrun[r] - newm);
        const float p0 = __expf(a0 - newm);
        const float p1 = __expf(a1 - newm);
        float rs = p0 + p1;
#pragma unroll
        for (int off = 1; off < 16; off <<= 1)
          rs += __shfl_xor(rs, off, 32);
        lsum[r] = lsum[r] * corr + rs;
        mrun[r] = newm;
#pragma unroll
        for (int nt = 0; nt < 4; ++nt) o[nt].f[r] *= corr;
        Plds[wave][r + hi * 8][ln]      = (bf16)p0;
        Plds[wave][r + hi * 8][16 + ln] = (bf16)p1;
      }
      asm volatile("s_wait_dscnt 0" ::: "memory");   // P tile visible to wave
      // O += P(16x32) * V(32x16) per dh tile
      Frag16 pa;
#pragma unroll
      for (int p = 0; p < 8; ++p) {
        int k2 = 2 * p + hi * 16;
        pa.u[p] = *reinterpret_cast<const unsigned int*>(&Plds[wave][ln][k2]);
      }
#pragma unroll
      for (int nt = 0; nt < 4; ++nt) {
        Frag16 vb;   // B frag: element e -> key = j0 + e + 16*hi, n = dh
        const bf16* vp = &Vt[nt * 16 + ln][j0 + hi * 16];
        *reinterpret_cast<uint4*>(&vb.u[0]) = *reinterpret_cast<const uint4*>(vp);
        *reinterpret_cast<uint4*>(&vb.u[4]) = *reinterpret_cast<const uint4*>(vp + 8);
        o[nt].v = wmma_bf16f32(pa.v, vb.v, o[nt].v);
      }
    }
    // normalize + write head-concat output [B,T,C] bf16
#pragma unroll
    for (int r = 0; r < 8; ++r) {
      const float inv = 1.0f / lsum[r];
      const int trow = rowbase + r + hi * 8;
#pragma unroll
      for (int nt = 0; nt < 4; ++nt) {
        const int col = h * DHD + nt * 16 + ln;
        att[((size_t)b * TSEQ + trow) * CDIM + col] = (bf16)(o[nt].f[r] * inv);
      }
    }
  }
}

// ---------------------------------------------------------------------------
// LM head (N=65, ragged -> VALU) + loss
// ---------------------------------------------------------------------------
__global__ __launch_bounds__(128)
void k_logits(const float* __restrict__ x, const float* __restrict__ W,
              const float* __restrict__ b, float* __restrict__ out) {
  __shared__ float row[CDIM];
  const int m = blockIdx.x, tid = threadIdx.x;
  for (int i = tid; i < CDIM; i += 128) row[i] = x[(size_t)m * CDIM + i];
  __syncthreads();
  if (tid < VOC) {
    float acc = b[tid];
    for (int c = 0; c < CDIM; ++c) acc += row[c] * W[c * VOC + tid];
    out[(size_t)m * VOC + tid] = acc;
  }
}

__global__ void k_zero(float* p) { if (threadIdx.x == 0) p[0] = 0.f; }

__global__ __launch_bounds__(256)
void k_loss(const float* __restrict__ logits, const int* __restrict__ tgt,
            float* __restrict__ loss) {
  __shared__ float red[256];
  const int m = blockIdx.x * 256 + threadIdx.x;
  const float* lp = logits + (size_t)m * VOC;
  float mx = -1e30f;
  for (int v = 0; v < VOC; ++v) mx = fmaxf(mx, lp[v]);
  float s = 0.f;
  for (int v = 0; v < VOC; ++v) s += __expf(lp[v] - mx);
  float l = -(lp[tgt[m]] - mx - __logf(s));
  red[threadIdx.x] = l;
  __syncthreads();
  for (int st = 128; st > 0; st >>= 1) {
    if (threadIdx.x < st) red[threadIdx.x] += red[threadIdx.x + st];
    __syncthreads();
  }
  if (threadIdx.x == 0) atomicAdd(loss, red[0] * (1.0f / MTOK));
}

// ---------------------------------------------------------------------------
extern "C" void kernel_launch(void* const* d_in, const int* in_sizes, int n_in,
                              void* d_out, int out_size, void* d_ws, size_t ws_size,
                              hipStream_t stream) {
  const int*   idx     = (const int*)d_in[0];
  const int*   target  = (const int*)d_in[1];
  const float* tok_emb = (const float*)d_in[2];
  const float* pos_emb = (const float*)d_in[3];
  const float* Wq      = (const float*)d_in[4];
  const float* Wk      = (const float*)d_in[5];
  const float* Wv      = (const float*)d_in[6];
  const float* Wo      = (const float*)d_in[7];
  const float* bo      = (const float*)d_in[8];
  const float* ln1_g   = (const float*)d_in[9];
  const float* ln1_b   = (const float*)d_in[10];
  const float* ln2_g   = (const float*)d_in[11];
  const float* ln2_b   = (const float*)d_in[12];
  const float* W1      = (const float*)d_in[13];
  const float* b1      = (const float*)d_in[14];
  const float* W2      = (const float*)d_in[15];
  const float* b2      = (const float*)d_in[16];
  const float* lnf_g   = (const float*)d_in[17];
  const float* lnf_b   = (const float*)d_in[18];
  const float* Wlm     = (const float*)d_in[19];
  const float* blm     = (const float*)d_in[20];

  float* logits = (float*)d_out;
  float* loss   = logits + (size_t)MTOK * VOC;

  size_t off = 0;
  auto carve = [&](size_t bytes) {
    void* p = (char*)d_ws + off;
    off += (bytes + 255) & ~(size_t)255;
    return p;
  };
  float* xf   = (float*)carve((size_t)MTOK * CDIM * 4);
  float* xlnf = (float*)carve((size_t)MTOK * CDIM * 4);
  bf16*  hb   = (bf16*)carve((size_t)MTOK * CDIM * 2);
  bf16*  qkvb = (bf16*)carve((size_t)MTOK * 3 * CDIM * 2);
  bf16*  attb = (bf16*)carve((size_t)MTOK * CDIM * 2);
  bf16*  mlpb = (bf16*)carve((size_t)MTOK * 4 * CDIM * 2);
  bf16*  wqkv = (bf16*)carve((size_t)NL * CDIM * 3 * CDIM * 2);
  bf16*  wo   = (bf16*)carve((size_t)NL * CDIM * CDIM * 2);
  bf16*  w1   = (bf16*)carve((size_t)NL * CDIM * 4 * CDIM * 2);
  bf16*  w2   = (bf16*)carve((size_t)NL * 4 * CDIM * CDIM * 2);
  (void)ws_size; (void)in_sizes; (void)n_in; (void)out_size;

  {  // weight packing (bf16)
    int n = NL * CDIM * 3 * CDIM;
    k_pack_qkv<<<(n + 255) / 256, 256, 0, stream>>>(Wq, Wk, Wv, wqkv, n);
    n = NL * CDIM * CDIM;
    k_convert_bf16<<<(n + 255) / 256, 256, 0, stream>>>(Wo, wo, n);
    n = NL * CDIM * 4 * CDIM;
    k_convert_bf16<<<(n + 255) / 256, 256, 0, stream>>>(W1, w1, n);
    k_convert_bf16<<<(n + 255) / 256, 256, 0, stream>>>(W2, w2, n);
  }
  {
    int n = MTOK * CDIM;
    k_embed<<<(n + 255) / 256, 256, 0, stream>>>(idx, tok_emb, pos_emb, xf);
  }

  for (int l = 0; l < NL; ++l) {
    k_layernorm<<<MTOK, 128, 0, stream>>>(xf, ln1_g + l * CDIM, ln1_b + l * CDIM,
                                          nullptr, hb);
    dim3 gq(MTOK / BM, (3 * CDIM) / BN);
    k_gemm_bf16<<<gq, 256, 0, stream>>>(hb, wqkv + (size_t)l * CDIM * 3 * CDIM,
                                        nullptr, nullptr, nullptr, qkvb,
                                        MTOK, 3 * CDIM, CDIM, 0);
    k_attention<<<(MTOK / TSEQ) * NH, 256, 0, stream>>>(qkvb, attb);
    dim3 go(MTOK / BM, CDIM / BN);
    k_gemm_bf16<<<go, 256, 0, stream>>>(attb, wo + (size_t)l * CDIM * CDIM,
                                        bo + l * CDIM, xf, xf, nullptr,
                                        MTOK, CDIM, CDIM, 0);
    k_layernorm<<<MTOK, 128, 0, stream>>>(xf, ln2_g + l * CDIM, ln2_b + l * CDIM,
                                          nullptr, hb);
    dim3 g1(MTOK / BM, (4 * CDIM) / BN);
    k_gemm_bf16<<<g1, 256, 0, stream>>>(hb, w1 + (size_t)l * CDIM * 4 * CDIM,
                                        b1 + l * 4 * CDIM, nullptr, nullptr, mlpb,
                                        MTOK, 4 * CDIM, CDIM, 1);
    dim3 g2(MTOK / BM, CDIM / BN);
    k_gemm_bf16<<<g2, 256, 0, stream>>>(mlpb, w2 + (size_t)l * 4 * CDIM * CDIM,
                                        b2 + l * CDIM, xf, xf, nullptr,
                                        MTOK, CDIM, 4 * CDIM, 0);
  }
  k_layernorm<<<MTOK, 128, 0, stream>>>(xf, lnf_g, lnf_b, xlnf, hb);
  k_logits<<<MTOK, 128, 0, stream>>>(xlnf, Wlm, blm, logits);
  k_zero<<<1, 1, 0, stream>>>(loss);
  k_loss<<<MTOK / 256, 256, 0, stream>>>(logits, target, loss);
}
